// Current_44435731645048
// MI455X (gfx1250) — compile-verified, matrix operand order checked by reading, not run on hardware
//
#include <hip/hip_runtime.h>
#include <hip/hip_bf16.h>

typedef __attribute__((ext_vector_type(16))) __bf16 v16bf;
typedef __attribute__((ext_vector_type(8)))  float  v8f;

union FragB { v16bf bf; uint4 u4[2]; };

#define HDIM 256
#define MSEQ 50
#define BT   64
#define PSTR 264   // padded bf16 row stride (528B, 16B aligned, bank-conflict-free)

__device__ __forceinline__ unsigned short f2bf(float f) {
    unsigned u = __float_as_uint(f);
    u += 0x7FFFu + ((u >> 16) & 1u);          // round-to-nearest-even
    return (unsigned short)(u >> 16);
}
__device__ __forceinline__ float bf2f(unsigned short b) {
    return __uint_as_float(((unsigned)b) << 16);
}
__device__ __forceinline__ float sigmoidf_(float x) {
    return 1.0f / (1.0f + __expf(-x));
}
// Async global->LDS copy of 16 bytes (CDNA5, tracked by ASYNCcnt).
__device__ __forceinline__ void async_g2l_b128(unsigned lds_byte_off, const void* gptr) {
    asm volatile("global_load_async_to_lds_b128 %0, %1, off"
                 :: "v"(lds_byte_off), "v"((unsigned long long)gptr) : "memory");
}
__device__ __forceinline__ void wait_asynccnt0() {
    asm volatile("s_wait_asynccnt 0" ::: "memory");
}

// ---------------------------------------------------------------------------
// Temporal bidirectional RNN (input_size=1, tanh) with the fusion dot product
// (hs_t . fW_half) fused into the step loop. One block = 64 sequences, one
// direction. Whh (pre-converted bf16) staged into LDS via async-DMA; h-state
// double-buffered in LDS -> single barrier per step. Matmul via WMMA bf16.
// dotbuf layout: [dir][t][seq] f32, dir in {0=fwd,1=bwd}, seq in [0,4096).
// ---------------------------------------------------------------------------
__global__ __launch_bounds__(256) void temporal_rnn_kernel(
    const float* __restrict__ x,
    const unsigned short* __restrict__ whhbf,   // bf16 (2,256,256): [dir][i][k]
    const float* __restrict__ Wih_f, const float* __restrict__ bih_f,
    const float* __restrict__ bhh_f,
    const float* __restrict__ Wih_b, const float* __restrict__ bih_b,
    const float* __restrict__ bhh_b,
    const float* __restrict__ fus_W,
    float* __restrict__ dotbuf)
{
    __shared__ __align__(16) unsigned short sW[HDIM * PSTR];      // Whh bf16 (row = out unit)
    __shared__ __align__(16) unsigned short sH2[2][BT * PSTR];    // h state bf16, ping-pong
    __shared__ float sWih[HDIM];
    __shared__ float sBias[HDIM];                                 // bih + bhh
    __shared__ float sFW[HDIM];                                   // fusion weights (this dir)
    __shared__ float sX[BT][MSEQ];

    const int tid = threadIdx.x;
    const int dir = blockIdx.y;
    const int n0  = blockIdx.x * BT;

    const unsigned short* whh = whhbf + dir * (HDIM * HDIM);
    const float* Wih = dir ? Wih_b : Wih_f;
    const float* bih = dir ? bih_b : bih_f;
    const float* bhh = dir ? bhh_b : bhh_f;

    // Async DMA Whh bf16 -> padded LDS (32 x b128 per thread, ASYNCcnt-tracked)
    {
        const unsigned lds_base = (unsigned)(size_t)(&sW[0]);
        for (int i = tid; i < (HDIM * HDIM) / 8; i += 256) {
            int r = i >> 5, ck = i & 31;              // row, 16B chunk in row
            unsigned loff = lds_base + (unsigned)(r * PSTR + ck * 8) * 2u;
            async_g2l_b128(loff, whh + r * HDIM + ck * 8);
        }
    }
    for (int i = tid; i < HDIM; i += 256) {
        sWih[i]  = Wih[i];
        sBias[i] = bih[i] + bhh[i];
        sFW[i]   = fus_W[dir * HDIM + i];
    }
    for (int i = tid; i < BT * PSTR; i += 256) sH2[0][i] = 0;     // h0 = 0
    for (int i = tid; i < BT * MSEQ; i += 256) {
        int j = i / MSEQ, t = i - j * MSEQ;
        int s = n0 + j;   // uv row: s<2048 -> x[...][t], s>=2048 -> x[...][50+t]
        sX[j][t] = x[(s & 2047) * 100 + ((s >= 2048) ? 50 : 0) + t];
    }
    wait_asynccnt0();
    __syncthreads();

    const int wave  = tid >> 5;
    const int lane  = tid & 31;
    const int lmod  = lane & 15;
    const int lhi   = lane >> 4;          // selects K half / M half per ISA layout
    const int ncol0 = wave * 32 + lmod;   // column of this lane's N-tile 0

    for (int step = 0; step < MSEQ; ++step) {
        const int t = dir ? (MSEQ - 1 - step) : step;
        const unsigned short* sHc = sH2[step & 1];        // read h_{t-1}
        unsigned short*       sHn = sH2[(step + 1) & 1];  // write h_t

        // C init: gi + (bih+bhh), per the 16x16 f32 C layout (VGPR r, lane l)
        v8f c[4][2];
        #pragma unroll
        for (int mt = 0; mt < 4; ++mt) {
            #pragma unroll
            for (int r = 0; r < 8; ++r) {
                int row  = mt * 16 + lhi * 8 + r;
                float xv = sX[row][t];
                c[mt][0][r] = xv * sWih[ncol0]      + sBias[ncol0];
                c[mt][1][r] = xv * sWih[ncol0 + 16] + sBias[ncol0 + 16];
            }
        }
        // K sweep: h(64x256) @ Whh^T(256x256) via 16x16x32 bf16 WMMA
        #pragma unroll
        for (int k0 = 0; k0 < HDIM; k0 += 32) {
            const int kb = k0 + lhi * 8;
            FragB a[4], b[2];
            #pragma unroll
            for (int mt = 0; mt < 4; ++mt) {
                const unsigned short* p = &sHc[(mt * 16 + lmod) * PSTR + kb];
                a[mt].u4[0] = *(const uint4*)(p);
                a[mt].u4[1] = *(const uint4*)(p + 16);
            }
            #pragma unroll
            for (int nt = 0; nt < 2; ++nt) {
                const unsigned short* p = &sW[(ncol0 + nt * 16) * PSTR + kb];
                b[nt].u4[0] = *(const uint4*)(p);
                b[nt].u4[1] = *(const uint4*)(p + 16);
            }
            #pragma unroll
            for (int mt = 0; mt < 4; ++mt)
                #pragma unroll
                for (int nt = 0; nt < 2; ++nt)
                    c[mt][nt] = __builtin_amdgcn_wmma_f32_16x16x32_bf16(
                        false, a[mt].bf, false, b[nt].bf,
                        (short)0, c[mt][nt], false, false);
        }
        // write h_t into the other buffer (no barrier needed before this)
        #pragma unroll
        for (int mt = 0; mt < 4; ++mt)
            #pragma unroll
            for (int nt = 0; nt < 2; ++nt)
                #pragma unroll
                for (int r = 0; r < 8; ++r) {
                    int row = mt * 16 + lhi * 8 + r;
                    int col = ncol0 + nt * 16;
                    sHn[row * PSTR + col] = f2bf(tanhf(c[mt][nt][r]));
                }
        __syncthreads();   // single barrier per step

        // fused fusion dot over ALL 256 threads: 4 threads per row, 64 cols each
        {
            const int drow = tid >> 2, dq = tid & 3;
            const uint4* hp = (const uint4*)&sHn[drow * PSTR + dq * 64];
            const float* fw = &sFW[dq * 64];
            float acc = 0.0f;
            #pragma unroll
            for (int v = 0; v < 8; ++v) {
                uint4 h4 = hp[v];
                const float* f8 = fw + v * 8;
                acc += bf2f((unsigned short)(h4.x))       * f8[0];
                acc += bf2f((unsigned short)(h4.x >> 16)) * f8[1];
                acc += bf2f((unsigned short)(h4.y))       * f8[2];
                acc += bf2f((unsigned short)(h4.y >> 16)) * f8[3];
                acc += bf2f((unsigned short)(h4.z))       * f8[4];
                acc += bf2f((unsigned short)(h4.z >> 16)) * f8[5];
                acc += bf2f((unsigned short)(h4.w))       * f8[6];
                acc += bf2f((unsigned short)(h4.w >> 16)) * f8[7];
            }
            acc += __shfl_xor(acc, 1);
            acc += __shfl_xor(acc, 2);
            if (dq == 0)
                dotbuf[(dir * MSEQ + t) * 4096 + n0 + drow] = acc;
        }
    }
}

// ---------------------------------------------------------------------------
// gate = sigmoid(dot_f + dot_b + fb); xg = x * gate (xg has x's layout since
// concat([gated[:b], gated[b:]], axis=2) reconstitutes (b,z,100)).
// ---------------------------------------------------------------------------
__global__ void gate_kernel(const float* __restrict__ x,
                            const float* __restrict__ dotbuf,
                            const float* __restrict__ fus_b,
                            float* __restrict__ xg)
{
    int idx = blockIdx.x * 256 + threadIdx.x;
    if (idx >= 4096 * MSEQ) return;
    int t = idx / 4096, s = idx - t * 4096;
    float d = dotbuf[t * 4096 + s] + dotbuf[(MSEQ + t) * 4096 + s] + fus_b[0];
    float g = sigmoidf_(d);
    int addr = (s & 2047) * 100 + ((s >= 2048) ? 50 : 0) + t;
    xg[addr] = x[addr] * g;
}

__global__ void cvt_bf16_kernel(const float* __restrict__ src,
                                unsigned short* __restrict__ dst, int n)
{
    int i = blockIdx.x * 256 + threadIdx.x;
    if (i < n) dst[i] = f2bf(src[i]);
}

// ---------------------------------------------------------------------------
// GRU input projection: gi[t][bb][g] = sum_e xin[bb][t][e]*Wih[g][e] + bih[g]
// WMMA-tiled GEMM, M=2048 (t*16+bb), N=768, K=E (zero-padded to 32).
// ---------------------------------------------------------------------------
__global__ __launch_bounds__(256) void gi_gemm_kernel(
    const float* __restrict__ xin,   // (16,128,E)
    const float* __restrict__ Wih,   // (768,E)
    const float* __restrict__ bih,   // (768)
    float* __restrict__ gi,          // (128,16,768)
    int E)
{
    __shared__ __align__(16) unsigned short sA[64 * 40];
    __shared__ __align__(16) unsigned short sB[64 * 40];
    const int tid  = threadIdx.x;
    const int m0   = blockIdx.x * 64;
    const int nb0  = blockIdx.y * 64;
    const int wave = tid >> 5, lane = tid & 31;
    const int lmod = lane & 15, lhi = lane >> 4;
    const int mt   = wave >> 1;          // 0..3
    const int ntb  = (wave & 1) * 2;     // 0 or 2

    v8f c[2];
    #pragma unroll
    for (int nt = 0; nt < 2; ++nt)
        #pragma unroll
        for (int r = 0; r < 8; ++r) c[nt][r] = 0.0f;

    const int kchunks = (E + 31) / 32;
    for (int kc = 0; kc < kchunks; ++kc) {
        const int k0 = kc * 32;
        for (int i = tid; i < 64 * 32; i += 256) {
            int mrow = i >> 5, kk = i & 31;
            int k = k0 + kk;
            int mg = m0 + mrow;               // = t*16 + bb
            int bb = mg & 15, tt = mg >> 4;
            float v = (k < E) ? xin[(bb * 128 + tt) * E + k] : 0.0f;
            sA[mrow * 40 + kk] = f2bf(v);
        }
        for (int i = tid; i < 64 * 32; i += 256) {
            int nrow = i >> 5, kk = i & 31;
            int k = k0 + kk;
            float v = (k < E) ? Wih[(nb0 + nrow) * E + k] : 0.0f;
            sB[nrow * 40 + kk] = f2bf(v);
        }
        __syncthreads();
        FragB a, b;
        const unsigned short* pa = &sA[(mt * 16 + lmod) * 40 + lhi * 8];
        a.u4[0] = *(const uint4*)(pa);
        a.u4[1] = *(const uint4*)(pa + 16);
        #pragma unroll
        for (int nt = 0; nt < 2; ++nt) {
            const unsigned short* pb = &sB[((ntb + nt) * 16 + lmod) * 40 + lhi * 8];
            b.u4[0] = *(const uint4*)(pb);
            b.u4[1] = *(const uint4*)(pb + 16);
            c[nt] = __builtin_amdgcn_wmma_f32_16x16x32_bf16(
                false, a.bf, false, b.bf, (short)0, c[nt], false, false);
        }
        __syncthreads();
    }
    #pragma unroll
    for (int nt = 0; nt < 2; ++nt)
        #pragma unroll
        for (int r = 0; r < 8; ++r) {
            int mrow = m0 + mt * 16 + lhi * 8 + r;
            int ncol = nb0 + (ntb + nt) * 16 + lmod;
            gi[mrow * 768 + ncol] = c[nt][r] + bih[ncol];
        }
}

// ---------------------------------------------------------------------------
// GRU recurrence: one persistent block. Per step: gh = h @ Whh^T + bhh via
// WMMA (h bf16 in LDS, Whh bf16 streamed from L2, next-step gi prefetched),
// then gate math in f32.
// ---------------------------------------------------------------------------
__global__ __launch_bounds__(256) void gru_recur_kernel(
    const float* __restrict__ gi,            // (128,16,768)
    const unsigned short* __restrict__ whh,  // bf16 (768,256)
    const float* __restrict__ bhh,           // (768)
    const float* __restrict__ h0,            // (16,256)
    float* __restrict__ out,                 // (16,128,256)
    float* __restrict__ hT)                  // (16,256)
{
    __shared__ __align__(16) unsigned short sH[16 * PSTR];  // h bf16 (matmul input)
    __shared__ float sHf[16 * 256];                         // h f32 (gate math)
    __shared__ float sGh[16 * 768];                         // h @ Whh^T + bhh

    const int tid  = threadIdx.x;
    const int wave = tid >> 5, lane = tid & 31;
    const int lmod = lane & 15, lhi = lane >> 4;

    for (int i = tid; i < 16 * 256; i += 256) {
        int bb = i >> 8, j = i & 255;
        float v = h0[i];
        sHf[i] = v;
        sH[bb * PSTR + j] = f2bf(v);
    }
    __syncthreads();

    for (int t = 0; t < 128; ++t) {
        v8f c[6];                         // 6 N-tiles per wave (48 total)
        #pragma unroll
        for (int i = 0; i < 6; ++i) {
            float bv = bhh[(wave * 6 + i) * 16 + lmod];
            #pragma unroll
            for (int r = 0; r < 8; ++r) c[i][r] = bv;
        }
        #pragma unroll
        for (int k0 = 0; k0 < 256; k0 += 32) {
            const int kb = k0 + lhi * 8;
            FragB a;
            const unsigned short* pa = &sH[lmod * PSTR + kb];
            a.u4[0] = *(const uint4*)(pa);
            a.u4[1] = *(const uint4*)(pa + 16);
            #pragma unroll
            for (int i = 0; i < 6; ++i) {
                int nrow = (wave * 6 + i) * 16 + lmod;
                FragB b;
                const unsigned short* pb = &whh[nrow * 256 + kb];
                b.u4[0] = *(const uint4*)(pb);
                b.u4[1] = *(const uint4*)(pb + 16);
                c[i] = __builtin_amdgcn_wmma_f32_16x16x32_bf16(
                    false, a.bf, false, b.bf, (short)0, c[i], false, false);
            }
        }
        #pragma unroll
        for (int i = 0; i < 6; ++i)
            #pragma unroll
            for (int r = 0; r < 8; ++r) {
                int bb  = lhi * 8 + r;
                int col = (wave * 6 + i) * 16 + lmod;
                sGh[bb * 768 + col] = c[i][r];
            }
        __syncthreads();
        for (int u = tid; u < 16 * 256; u += 256) {
            int bb = u >> 8, j = u & 255;
            const float* g = &gi[(t * 16 + bb) * 768];
            if (t + 1 < 128)   // prefetch next step's gi rows into cache
                __builtin_prefetch(&gi[((t + 1) * 16 + bb) * 768 + j], 0, 1);
            float ir = g[j], iz = g[256 + j], inn = g[512 + j];
            float hr = sGh[bb * 768 + j];
            float hz = sGh[bb * 768 + 256 + j];
            float hn = sGh[bb * 768 + 512 + j];
            float rr = sigmoidf_(ir + hr);
            float zz = sigmoidf_(iz + hz);
            float nn = tanhf(inn + rr * hn);
            float hv = (1.0f - zz) * nn + zz * sHf[u];
            sHf[u] = hv;
            sH[bb * PSTR + j] = f2bf(hv);
            out[(bb * 128 + t) * 256 + j] = hv;
        }
        __syncthreads();
    }
    for (int i = tid; i < 16 * 256; i += 256) hT[i] = sHf[i];
}

// ---------------------------------------------------------------------------
extern "C" void kernel_launch(void* const* d_in, const int* in_sizes, int n_in,
                              void* d_out, int out_size, void* d_ws, size_t ws_size,
                              hipStream_t stream)
{
    const float* x     = (const float*)d_in[0];
    const float* h0    = (const float*)d_in[1];
    const float* tWihF = (const float*)d_in[2];
    const float* tWhhF = (const float*)d_in[3];
    const float* tBihF = (const float*)d_in[4];
    const float* tBhhF = (const float*)d_in[5];
    const float* tWihB = (const float*)d_in[6];
    const float* tWhhB = (const float*)d_in[7];
    const float* tBihB = (const float*)d_in[8];
    const float* tBhhB = (const float*)d_in[9];
    const float* fusW  = (const float*)d_in[10];
    const float* fusB  = (const float*)d_in[11];
    const float* gWih0 = (const float*)d_in[12];
    const float* gWhh0 = (const float*)d_in[13];
    const float* gBih0 = (const float*)d_in[14];
    const float* gBhh0 = (const float*)d_in[15];
    const float* gWih1 = (const float*)d_in[16];
    const float* gWhh1 = (const float*)d_in[17];
    const float* gBih1 = (const float*)d_in[18];
    const float* gBhh1 = (const float*)d_in[19];
    (void)in_sizes; (void)n_in; (void)out_size; (void)ws_size;

    float* out1 = (float*)d_out;                 // (16,128,256)
    float* hTs  = out1 + 16 * 128 * 256;         // (2,16,256)

    char* ws = (char*)d_ws;
    float* dotbuf = (float*)ws;  ws += (size_t)2 * MSEQ * 4096 * sizeof(float);
    float* xg     = (float*)ws;  ws += (size_t)16 * 128 * 100 * sizeof(float);
    float* gi     = (float*)ws;  ws += (size_t)2048 * 768 * sizeof(float);   // reused by both layers
    float* out0   = (float*)ws;  ws += (size_t)2048 * 256 * sizeof(float);
    unsigned short* whhb  = (unsigned short*)ws; ws += (size_t)768 * 256 * sizeof(unsigned short);
    unsigned short* twhhb = (unsigned short*)ws; ws += (size_t)2 * HDIM * HDIM * sizeof(unsigned short);

    // 0) one-time bf16 conversion of the temporal Whh weights (fwd+bwd)
    cvt_bf16_kernel<<<(HDIM * HDIM + 255) / 256, 256, 0, stream>>>(tWhhF, twhhb, HDIM * HDIM);
    cvt_bf16_kernel<<<(HDIM * HDIM + 255) / 256, 256, 0, stream>>>(tWhhB, twhhb + HDIM * HDIM, HDIM * HDIM);

    // 1) Temporal bidirectional RNN with fused fusion-dot (dominant stage)
    temporal_rnn_kernel<<<dim3(64, 2), 256, 0, stream>>>(
        x, twhhb, tWihF, tBihF, tBhhF, tWihB, tBihB, tBhhB, fusW, dotbuf);

    // 2) gate + scatter into (b,z,100) layout
    gate_kernel<<<(4096 * MSEQ + 255) / 256, 256, 0, stream>>>(x, dotbuf, fusB, xg);

    // 3) GRU layer 0
    cvt_bf16_kernel<<<(768 * 256 + 255) / 256, 256, 0, stream>>>(gWhh0, whhb, 768 * 256);
    gi_gemm_kernel<<<dim3(32, 12), 256, 0, stream>>>(xg, gWih0, gBih0, gi, 100);
    gru_recur_kernel<<<1, 256, 0, stream>>>(gi, whhb, gBhh0, h0, out0, hTs);

    // 4) GRU layer 1
    cvt_bf16_kernel<<<(768 * 256 + 255) / 256, 256, 0, stream>>>(gWhh1, whhb, 768 * 256);
    gi_gemm_kernel<<<dim3(32, 12), 256, 0, stream>>>(out0, gWih1, gBih1, gi, 256);
    gru_recur_kernel<<<1, 256, 0, stream>>>(gi, whhb, gBhh1, h0 + 16 * 256, out1, hTs + 16 * 256);
}
